// FBANetLayer_7550552506692
// MI455X (gfx1250) — matrix operand order repaired and checked by reading, not compiled, not run on hardware
//
#include <hip/hip_runtime.h>

// ---------------------------------------------------------------------------
// Swin block for MI455X (gfx1250, wave32, WMMA). All matmuls via
// v_wmma_f32_16x16x32_bf16 (f32 ref down-converted to bf16, f32 accum).
// GEMM uses a double-buffered LDS pipeline fed by
// GLOBAL_LOAD_ASYNC_TO_LDS_B128 (ASYNCcnt), overlapping copy(i+1) with
// WMMA compute(i).
// ---------------------------------------------------------------------------

typedef __attribute__((ext_vector_type(16))) __bf16 v16bf;
typedef __attribute__((ext_vector_type(8)))  float  v8f;

union U16 { uint4 u4[2]; v16bf v; };

__device__ __forceinline__ v8f v8f_zero() {
  v8f z = {0.f, 0.f, 0.f, 0.f, 0.f, 0.f, 0.f, 0.f};
  return z;
}

__device__ __forceinline__ v8f wmma_bf16(v16bf a, v16bf b, v8f c) {
  return __builtin_amdgcn_wmma_f32_16x16x32_bf16(
      /*neg_a=*/false, a, /*neg_b=*/false, b,
      /*c_mod=*/(short)0, c, /*reuse_a=*/false, /*reuse_b=*/false);
}

__device__ __forceinline__ unsigned short f2bf(float f) {
  union { float f; unsigned u; } x; x.f = f;
  unsigned r = x.u + 0x7FFFu + ((x.u >> 16) & 1u);   // round-to-nearest-even
  return (unsigned short)(r >> 16);
}

// Async global->LDS copy, 16B per lane. VDST = LDS byte offset (low 32 bits
// of the generic shared-memory address per the LDS aperture mapping).
__device__ __forceinline__ void async_copy_b128(void* lds_dst, const void* gsrc) {
  unsigned lds = (unsigned)(unsigned long long)lds_dst;
  asm volatile("global_load_async_to_lds_b128 %0, %1, off"
               :: "v"(lds), "v"(gsrc)
               : "memory");
}
__device__ __forceinline__ void wait_asynccnt0() {
  asm volatile("s_wait_asynccnt 0x0" ::: "memory");
}

// ---------------------------------------------------------------------------
// Constants (from reference)
// ---------------------------------------------------------------------------
#define CH   128          // image H
#define CW   128          // image W
#define CWH  8            // window H
#define CWW  8            // window W
#define CSH  4            // shift
#define CDIM 512
#define CHEADS 16
#define CHD  32
#define CHID 2048
#define CNH  16           // windows per dim
#define CNWIN 256
#define CWL  64
#define CTOK (CH*CW)      // 16384
#define CSCALE 0.1767766953f   // 32^-0.5

// ---------------------------------------------------------------------------
// f32 -> bf16 conversion
// ---------------------------------------------------------------------------
__global__ void cvt_f32_bf16(const float* __restrict__ src,
                             unsigned short* __restrict__ dst, int n) {
  int i = blockIdx.x * blockDim.x + threadIdx.x;
  if (i < n) dst[i] = f2bf(src[i]);
}

// ---------------------------------------------------------------------------
// LayerNorm over 512-wide rows; SHIFT=true also applies cyclic shift (-4,-4)
// + window partition so output row order = (win, in-window-pos). bf16 out.
// One wave per row, 8 rows per 256-thread block.
// ---------------------------------------------------------------------------
template <bool SHIFT>
__global__ __launch_bounds__(256) void ln_kernel(const float* __restrict__ x,
                                                 const float* __restrict__ w,
                                                 const float* __restrict__ b,
                                                 unsigned short* __restrict__ out) {
  int r    = blockIdx.x * 8 + (threadIdx.x >> 5);   // 0..16383
  int lane = threadIdx.x & 31;

  int src;
  if (SHIFT) {
    int win = r >> 6, l = r & 63;
    int wi = win >> 4, wj = win & 15, ai = l >> 3, aj = l & 7;
    int sr = (wi * 8 + ai + CSH) & (CH - 1);
    int sc = (wj * 8 + aj + CSH) & (CW - 1);
    src = sr * CW + sc;
  } else {
    src = r;
  }
  const float* xr = x + (size_t)src * CDIM;

  float4 vb[4];
  float s = 0.f, ss = 0.f;
#pragma unroll
  for (int j = 0; j < 4; ++j) {
    float4 v = ((const float4*)xr)[lane + 32 * j];
    vb[j] = v;
    s  += v.x + v.y + v.z + v.w;
    ss += v.x * v.x + v.y * v.y + v.z * v.z + v.w * v.w;
  }
#pragma unroll
  for (int off = 16; off >= 1; off >>= 1) {
    s  += __shfl_xor(s,  off, 32);
    ss += __shfl_xor(ss, off, 32);
  }
  float mu   = s * (1.f / CDIM);
  float var  = ss * (1.f / CDIM) - mu * mu;
  float rstd = rsqrtf(var + 1e-5f);

  unsigned short* orow = out + (size_t)r * CDIM;
#pragma unroll
  for (int j = 0; j < 4; ++j) {
    float vv[4] = {vb[j].x, vb[j].y, vb[j].z, vb[j].w};
#pragma unroll
    for (int c = 0; c < 4; ++c) {
      int col = (lane + 32 * j) * 4 + c;
      orow[col] = f2bf((vv[c] - mu) * rstd * w[col] + b[col]);
    }
  }
}

// ---------------------------------------------------------------------------
// Generic bf16 WMMA GEMM: C(M x N) = A(M x K) * B(N x K)^T, epilogue functor.
// Block tile 128x128, 8 waves, each wave 32x64 (2 M-tiles x 4 N-tiles).
// Double-buffered LDS staging via async-to-LDS b128 copies (ASYNCcnt):
// copies for K-step i+1 overlap WMMA compute of K-step i.
// ---------------------------------------------------------------------------
template <class Epi>
__global__ __launch_bounds__(256) void gemm_bf16(
    const unsigned short* __restrict__ A,
    const unsigned short* __restrict__ B,
    int K, Epi epi) {
  __shared__ uint4 As4[2][512];   // 2 x (128 rows x 32 cols bf16)
  __shared__ uint4 Bs4[2][512];   // 2 x (128 cols x 32 k bf16)

  int mBase = blockIdx.y * 128;
  int nBase = blockIdx.x * 128;
  int tid   = threadIdx.x;
  int lane  = tid & 31, lhalf = lane & 15, hi = lane >> 4;
  int wave  = tid >> 5;
  int wm = wave >> 1;     // 0..3 -> 32 rows each
  int wn = wave & 1;      // 0..1 -> 64 cols each

  v8f acc[2][4];
#pragma unroll
  for (int mt = 0; mt < 2; ++mt)
#pragma unroll
    for (int nt = 0; nt < 4; ++nt) acc[mt][nt] = v8f_zero();

  int lin  = tid * 8;
  int r0 = lin >> 5,  c0 = lin & 31;
  int lin2 = (tid + 256) * 8;
  int r1 = lin2 >> 5, c1 = lin2 & 31;

  const unsigned short* Arow0 = A + (size_t)(mBase + r0) * K + c0;
  const unsigned short* Arow1 = A + (size_t)(mBase + r1) * K + c1;
  const unsigned short* Brow0 = B + (size_t)(nBase + r0) * K + c0;
  const unsigned short* Brow1 = B + (size_t)(nBase + r1) * K + c1;

  auto issue = [&](int buf, int kb) {
    async_copy_b128(&As4[buf][tid],       Arow0 + kb);
    async_copy_b128(&As4[buf][tid + 256], Arow1 + kb);
    async_copy_b128(&Bs4[buf][tid],       Brow0 + kb);
    async_copy_b128(&Bs4[buf][tid + 256], Brow1 + kb);
  };

  int steps = K >> 5;
  issue(0, 0);                              // prologue: stage K-step 0

  for (int i = 0; i < steps; ++i) {
    int cur = i & 1;
    // own copies for buf[cur] are the only outstanding async ops
    wait_asynccnt0();
    // all waves: buf[cur] fully written, and everyone is done reading
    // buf[cur^1] (their previous compute) -> safe to overwrite it next
    __syncthreads();
    if (i + 1 < steps) {
      issue(cur ^ 1, (i + 1) * 32);         // overlap copy(i+1) with compute(i)
      if (i + 2 < steps) {                  // keep L2 primed two tiles ahead
        __builtin_prefetch(Arow0 + (i + 2) * 32, 0, 0);
        __builtin_prefetch(Brow0 + (i + 2) * 32, 0, 0);
      }
    }

    const uint4* Ab = As4[cur];
    const uint4* Bb = Bs4[cur];
    U16 fa[2], fb[4];
    int k0  = hi ? 8 : 0;    // A fragment: halves {k0..k0+7, k0+16..k0+23}
    int kb0 = hi ? 8 : 0;    // B fragment: halves {kb0..kb0+7, kb0+16..kb0+23}
#pragma unroll
    for (int mt = 0; mt < 2; ++mt) {
      int row = wm * 32 + mt * 16 + lhalf;
      fa[mt].u4[0] = Ab[row * 4 + (k0 >> 3)];
      fa[mt].u4[1] = Ab[row * 4 + (k0 >> 3) + 2];
    }
#pragma unroll
    for (int nt = 0; nt < 4; ++nt) {
      int n = wn * 64 + nt * 16 + lhalf;
      fb[nt].u4[0] = Bb[n * 4 + (kb0 >> 3)];
      fb[nt].u4[1] = Bb[n * 4 + (kb0 >> 3) + 2];
    }
#pragma unroll
    for (int mt = 0; mt < 2; ++mt)
#pragma unroll
      for (int nt = 0; nt < 4; ++nt)
        acc[mt][nt] = wmma_bf16(fa[mt].v, fb[nt].v, acc[mt][nt]);
    // no trailing barrier: next iteration's wait+barrier provides it
  }

#pragma unroll
  for (int mt = 0; mt < 2; ++mt)
#pragma unroll
    for (int nt = 0; nt < 4; ++nt)
#pragma unroll
      for (int v = 0; v < 8; ++v) {
        int row = mBase + wm * 32 + mt * 16 + v + hi * 8;
        int col = nBase + wn * 64 + nt * 16 + lhalf;
        epi(row, col, acc[mt][nt][v]);
      }
}

// ---------------------------------------------------------------------------
// Epilogues
// ---------------------------------------------------------------------------
struct EpiQKV {
  const float* bias;
  unsigned short *Q, *Kd, *V;
  __device__ void operator()(int m, int e, float acc) const {
    float val = acc + bias[e];
    int sec = e >> 9, rr = e & 511, h = rr >> 5, d = rr & 31;
    int win = m >> 6, l = m & 63;
    size_t wh = (size_t)(win * CHEADS + h);
    if (sec == 0)      Q [(wh * CWL + l) * CHD + d] = f2bf(val * CSCALE);
    else if (sec == 1) Kd[(wh * CWL + l) * CHD + d] = f2bf(val);
    else               V [(wh * CHD + d) * CWL + l] = f2bf(val);  // transposed
  }
};

struct EpiProj {   // window-reverse + reverse cyclic shift + residual
  const float* bias;
  const float* skip;
  float* h1;
  __device__ void operator()(int m, int e, float acc) const {
    int win = m >> 6, l = m & 63;
    int wi = win >> 4, wj = win & 15, ai = l >> 3, aj = l & 7;
    int dr = (wi * 8 + ai + CSH) & (CH - 1);
    int dc = (wj * 8 + aj + CSH) & (CW - 1);
    size_t dst = (size_t)(dr * CW + dc) * CDIM + e;
    h1[dst] = skip[dst] + acc + bias[e];
  }
};

struct EpiMlp1 {   // bias + tanh-approx GELU -> bf16
  const float* bias;
  unsigned short* hid;
  __device__ void operator()(int m, int e, float acc) const {
    float x = acc + bias[e];
    float t = tanhf(0.7978845608f * (x + 0.044715f * x * x * x));
    hid[(size_t)m * CHID + e] = f2bf(0.5f * x * (1.f + t));
  }
};

struct EpiMlp2 {   // bias, out = m + m
  const float* bias;
  float* out;
  __device__ void operator()(int m, int e, float acc) const {
    out[(size_t)m * CDIM + e] = 2.f * (acc + bias[e]);
  }
};

// ---------------------------------------------------------------------------
// Windowed multi-head attention. One block per (win, head), 4 waves.
// Wave w handles query rows [16w, 16w+16). K dim = HD = 32 = one bf16 WMMA.
// ---------------------------------------------------------------------------
__global__ __launch_bounds__(128) void attn_kernel(
    const unsigned short* __restrict__ Qh,
    const unsigned short* __restrict__ Kh,
    const unsigned short* __restrict__ Vt,
    const float* __restrict__ rpb,
    unsigned short* __restrict__ attn_o) {
  int win  = blockIdx.x >> 4;
  int h    = blockIdx.x & 15;
  int wave = threadIdx.x >> 5;
  int lane = threadIdx.x & 31;
  int lhalf = lane & 15, hi = lane >> 4;

  __shared__ unsigned short Pls[4][16][64];   // per-wave P tile (bf16)

  size_t wh = (size_t)(win * CHEADS + h);
  const unsigned short* Qb = Qh + wh * CWL * CHD;
  const unsigned short* Kb = Kh + wh * CWL * CHD;
  const unsigned short* Vb = Vt + wh * CHD * CWL;

  // Q fragment (A: 16x32, rows = query positions, K = head dim)
  int qrow = wave * 16 + lhalf;
  int k0   = hi ? 8 : 0;
  U16 aQ;
  aQ.u4[0] = *(const uint4*)(Qb + qrow * CHD + k0);
  aQ.u4[1] = *(const uint4*)(Qb + qrow * CHD + k0 + 16);

  // S = Q * K^T : 4 column tiles of 16
  v8f S[4];
#pragma unroll
  for (int t = 0; t < 4; ++t) {
    int kn = t * 16 + lhalf;          // key position = B column
    int kb = hi ? 8 : 0;              // B fragment K halves
    U16 bK;
    bK.u4[0] = *(const uint4*)(Kb + kn * CHD + kb);
    bK.u4[1] = *(const uint4*)(Kb + kn * CHD + kb + 16);
    S[t] = wmma_bf16(aQ.v, bK.v, v8f_zero());
  }

  // relative-position bias + shifted-window mask (computed analytically)
  int wi = win >> 4, wj = win & 15;
#pragma unroll
  for (int t = 0; t < 4; ++t)
#pragma unroll
    for (int v = 0; v < 8; ++v) {
      int ql = wave * 16 + v + hi * 8;
      int kl = t * 16 + lhalf;
      int ai = ql >> 3, aj = ql & 7, bi = kl >> 3, bj = kl & 7;
      int rpi = (ai - bi + CWH - 1) * (2 * CWW - 1) + (aj - bj + CWW - 1);
      float bias = rpb[rpi * CHEADS + h];
      int rq = wi * 8 + ai, cq = wj * 8 + aj;
      int rk = wi * 8 + bi, ck = wj * 8 + bj;
      int gq = (rq < CH - CWH ? 0 : (rq < CH - CSH ? 1 : 2)) * 3 +
               (cq < CW - CWW ? 0 : (cq < CW - CSH ? 1 : 2));
      int gk = (rk < CH - CWH ? 0 : (rk < CH - CSH ? 1 : 2)) * 3 +
               (ck < CW - CWW ? 0 : (ck < CW - CSH ? 1 : 2));
      S[t][v] += bias + (gq != gk ? -100.f : 0.f);
    }

  // softmax along keys (row = (v, lane-half); 64 vals over 4 tiles x 16 lanes)
#pragma unroll
  for (int v = 0; v < 8; ++v) {
    float mx = fmaxf(fmaxf(S[0][v], S[1][v]), fmaxf(S[2][v], S[3][v]));
#pragma unroll
    for (int off = 8; off >= 1; off >>= 1) mx = fmaxf(mx, __shfl_xor(mx, off, 32));
    float sum = 0.f;
#pragma unroll
    for (int t = 0; t < 4; ++t) {
      float e = __expf(S[t][v] - mx);
      S[t][v] = e;
      sum += e;
    }
#pragma unroll
    for (int off = 8; off >= 1; off >>= 1) sum += __shfl_xor(sum, off, 32);
    float rinv = __frcp_rn(sum);
#pragma unroll
    for (int t = 0; t < 4; ++t) S[t][v] *= rinv;
  }

  // stage P (C-layout) -> LDS so we can re-read in A-fragment layout
#pragma unroll
  for (int t = 0; t < 4; ++t)
#pragma unroll
    for (int v = 0; v < 8; ++v)
      Pls[wave][v + hi * 8][t * 16 + lhalf] = f2bf(S[t][v]);
  __syncthreads();

  // O = P * V : K = 64 (2 steps of 32), N = 32 (2 tiles)
  v8f O[2] = {v8f_zero(), v8f_zero()};
#pragma unroll
  for (int s = 0; s < 2; ++s) {
    int pk0 = s * 32 + (hi ? 8 : 0);
    U16 aP;
    aP.u4[0] = *(const uint4*)&Pls[wave][lhalf][pk0];
    aP.u4[1] = *(const uint4*)&Pls[wave][lhalf][pk0 + 16];
#pragma unroll
    for (int u = 0; u < 2; ++u) {
      int d   = u * 16 + lhalf;          // head-dim column
      int vkb = s * 32 + (hi ? 8 : 0);   // key index halves
      U16 bV;
      bV.u4[0] = *(const uint4*)(Vb + d * CWL + vkb);
      bV.u4[1] = *(const uint4*)(Vb + d * CWL + vkb + 16);
      O[u] = wmma_bf16(aP.v, bV.v, O[u]);
    }
  }

  // write (win, l, h*32+d) as bf16 for proj GEMM input
#pragma unroll
  for (int u = 0; u < 2; ++u)
#pragma unroll
    for (int v = 0; v < 8; ++v) {
      int row = wave * 16 + v + hi * 8;
      int col = u * 16 + lhalf;
      attn_o[(size_t)(win * CWL + row) * CDIM + h * CHD + col] = f2bf(O[u][v]);
    }
}

// ---------------------------------------------------------------------------
// Host-side orchestration
// ---------------------------------------------------------------------------
extern "C" void kernel_launch(void* const* d_in, const int* in_sizes, int n_in,
                              void* d_out, int out_size, void* d_ws, size_t ws_size,
                              hipStream_t stream) {
  (void)in_sizes; (void)n_in; (void)out_size; (void)ws_size;

  const float* x      = (const float*)d_in[0];
  const float* ln1_w  = (const float*)d_in[1];
  const float* ln1_b  = (const float*)d_in[2];
  const float* ln2_w  = (const float*)d_in[3];
  const float* ln2_b  = (const float*)d_in[4];
  const float* qkv_w  = (const float*)d_in[5];
  const float* qkv_b  = (const float*)d_in[6];
  const float* proj_w = (const float*)d_in[7];
  const float* proj_b = (const float*)d_in[8];
  const float* rpb    = (const float*)d_in[9];
  const float* w1     = (const float*)d_in[10];
  const float* b1     = (const float*)d_in[11];
  const float* w2     = (const float*)d_in[12];
  const float* b2     = (const float*)d_in[13];
  float* out = (float*)d_out;

  char* ws = (char*)d_ws;
  size_t off = 0;
  auto take = [&](size_t bytes) -> char* {
    char* p = ws + off;
    off = (off + bytes + 255) & ~(size_t)255;
    return p;
  };

  const size_t NQKVW = 3 * CDIM * CDIM;     // 786432
  const size_t NPROJ = CDIM * CDIM;         // 262144
  const size_t NW1   = CHID * CDIM;         // 1048576
  const size_t NW2   = CDIM * CHID;         // 1048576
  const size_t NACT  = (size_t)CTOK * CDIM; // 8388608
  const size_t NHEAD = (size_t)CNWIN * CHEADS * CWL * CHD;  // 8388608

  unsigned short* qkv_wb = (unsigned short*)take(NQKVW * 2);
  unsigned short* proj_wb= (unsigned short*)take(NPROJ * 2);
  unsigned short* w1b    = (unsigned short*)take(NW1 * 2);
  unsigned short* w2b    = (unsigned short*)take(NW2 * 2);
  unsigned short* xw     = (unsigned short*)take(NACT * 2);
  unsigned short* Qh     = (unsigned short*)take(NHEAD * 2);
  unsigned short* Kh     = (unsigned short*)take(NHEAD * 2);
  unsigned short* Vt     = (unsigned short*)take(NHEAD * 2);
  unsigned short* attn_o = (unsigned short*)take(NACT * 2);
  float*          h1     = (float*)take(NACT * 4);
  // reuse dead regions:
  unsigned short* hln    = xw;   // xw dead after QKV GEMM
  unsigned short* hidden = Qh;   // Qh..attn_o (4 * 16.8MB = 67.1MB) >= hidden

  // 1) convert weights to bf16
  cvt_f32_bf16<<<(int)((NQKVW + 255) / 256), 256, 0, stream>>>(qkv_w, qkv_wb, (int)NQKVW);
  cvt_f32_bf16<<<(int)((NPROJ + 255) / 256), 256, 0, stream>>>(proj_w, proj_wb, (int)NPROJ);
  cvt_f32_bf16<<<(int)((NW1 + 255) / 256), 256, 0, stream>>>(w1, w1b, (int)NW1);
  cvt_f32_bf16<<<(int)((NW2 + 255) / 256), 256, 0, stream>>>(w2, w2b, (int)NW2);

  // 2) LN1 + cyclic shift + window partition
  ln_kernel<true><<<CTOK / 8, 256, 0, stream>>>(x, ln1_w, ln1_b, xw);

  // 3) QKV GEMM (16384 x 512 x 1536), scatter into head layouts
  gemm_bf16<EpiQKV><<<dim3(1536 / 128, CTOK / 128), 256, 0, stream>>>(
      xw, qkv_wb, CDIM, EpiQKV{qkv_b, Qh, Kh, Vt});

  // 4) windowed attention
  attn_kernel<<<CNWIN * CHEADS, 128, 0, stream>>>(Qh, Kh, Vt, rpb, attn_o);

  // 5) proj GEMM + window reverse + reverse shift + residual
  gemm_bf16<EpiProj><<<dim3(CDIM / 128, CTOK / 128), 256, 0, stream>>>(
      attn_o, proj_wb, CDIM, EpiProj{proj_b, x, h1});

  // 6) LN2
  ln_kernel<false><<<CTOK / 8, 256, 0, stream>>>(h1, ln2_w, ln2_b, hln);

  // 7) MLP fc1 + GELU
  gemm_bf16<EpiMlp1><<<dim3(CHID / 128, CTOK / 128), 256, 0, stream>>>(
      hln, w1b, CDIM, EpiMlp1{b1, hidden});

  // 8) MLP fc2, out = m + m
  gemm_bf16<EpiMlp2><<<dim3(CDIM / 128, CTOK / 128), 256, 0, stream>>>(
      hidden, w2b, CHID, EpiMlp2{b2, out});
}
